// energy_latencyGNN_50_41446434406429
// MI455X (gfx1250) — compile-verified, matrix operand order checked by reading, not run on hardware
//
#include <hip/hip_runtime.h>
#include <math.h>

typedef __attribute__((ext_vector_type(16))) __bf16 v16bf;
typedef __attribute__((ext_vector_type(8)))  float  v8f;
typedef __attribute__((ext_vector_type(4)))  unsigned int v4u;
typedef __attribute__((ext_vector_type(8)))  int v8i;
typedef __attribute__((ext_vector_type(4)))  int v4i;

#define BB    8192
#define NND   100
#define EED   1600
#define EMBD  5
#define HROW  500      // NND*EMBD
#define DCOL  2600
#define XCOL  3100
#define KP    3104     // XCOL padded to multiple of 32
#define NCH   (KP / 32)  // 97 K-chunks
#define NH1   128
#define NH2   128
#define NH3   64
#define NH4P  16       // 2 padded to one WMMA N tile

// ---------------- CSR construction (graph is batch-invariant) ----------------
__global__ void k_zero_i32(int* p, int n) {
    int i = blockIdx.x * blockDim.x + threadIdx.x;
    if (i < n) p[i] = 0;
}
__global__ void k_csr_count(const int* __restrict__ dst, int* __restrict__ cnt) {
    int e = blockIdx.x * blockDim.x + threadIdx.x;
    if (e < EED) atomicAdd(&cnt[dst[e]], 1);
}
__global__ void k_csr_scan(const int* __restrict__ cnt, int* __restrict__ rowptr,
                           int* __restrict__ fill) {
    if (blockIdx.x == 0 && threadIdx.x == 0) {
        int s = 0;
        for (int n = 0; n < NND; ++n) { rowptr[n] = s; fill[n] = s; s += cnt[n]; }
        rowptr[NND] = s;
    }
}
__global__ void k_csr_fill(const int* __restrict__ src, const int* __restrict__ dst,
                           int* __restrict__ fill, int* __restrict__ cols) {
    int e = blockIdx.x * blockDim.x + threadIdx.x;
    if (e < EED) {
        int p = atomicAdd(&fill[dst[e]], 1);
        cols[p] = src[e];
    }
}

// ---------------- fused gated-RGCN layer (gather + W + gate + lrelu) ----------------
__global__ void k_rgcn(const float* __restrict__ hin, float* __restrict__ hout,
                       const int* __restrict__ rowptr, const int* __restrict__ cols,
                       const float* __restrict__ W, const float* __restrict__ bias,
                       const float* __restrict__ G, const float* __restrict__ gb,
                       int in_dim) {
    int idx = blockIdx.x * blockDim.x + threadIdx.x;
    if (idx >= BB * NND) return;
    int b = idx / NND, n = idx % NND;
    float agg[EMBD] = {0.f, 0.f, 0.f, 0.f, 0.f};
    const int p0 = rowptr[n], p1 = rowptr[n + 1];
    const float* hb = hin + (size_t)b * NND * in_dim;
    for (int p = p0; p < p1; ++p) {
        const float* hr = hb + cols[p] * in_dim;
        #pragma unroll 5
        for (int c = 0; c < in_dim; ++c) agg[c] += hr[c];
    }
    float h2[EMBD];
    #pragma unroll
    for (int j = 0; j < EMBD; ++j) {
        float s = bias[j];
        for (int c = 0; c < in_dim; ++c) s += agg[c] * W[c * EMBD + j];
        h2[j] = s;
    }
    float* orow = hout + (size_t)idx * EMBD;
    #pragma unroll
    for (int j = 0; j < EMBD; ++j) {
        float s = gb[j];
        #pragma unroll
        for (int c = 0; c < EMBD; ++c) s += h2[c] * G[c * EMBD + j];
        float g = 1.0f / (1.0f + __expf(-s));
        float v = g * h2[j];
        orow[j] = v > 0.f ? v : 0.01f * v;
    }
}

// ---------------- pack x = concat(h, d) -> bf16, K padded to 3104 ----------------
__global__ void k_pack(const float* __restrict__ h, const float* __restrict__ d,
                       __bf16* __restrict__ X) {
    int idx = blockIdx.x * blockDim.x + threadIdx.x;
    if (idx >= BB * KP) return;
    int b = idx / KP, k = idx % KP;
    float v = 0.f;
    if (k < HROW)      v = h[(size_t)b * HROW + k];
    else if (k < XCOL) v = d[(size_t)b * DCOL + (k - HROW)];
    X[idx] = (__bf16)v;
}

// ---------------- weight transpose + bf16 convert: Wt[n*Kpad+k] = W[k*Nreal+n] ----------------
__global__ void k_wt(const float* __restrict__ W, __bf16* __restrict__ Wt,
                     int K, int Kpad, int Nreal, int Npad) {
    int idx = blockIdx.x * blockDim.x + threadIdx.x;
    if (idx >= Npad * Kpad) return;
    int n = idx / Kpad, k = idx % Kpad;
    float v = (k < K && n < Nreal) ? W[(size_t)k * Nreal + n] : 0.f;
    Wt[idx] = (__bf16)v;
}

// ---------------- helpers ----------------
__device__ __forceinline__ v16bf ldv16(const __bf16* p) {
    return *reinterpret_cast<const v16bf*>(p);
}

// TDM: one TENSOR_LOAD_TO_LDS of a rows x cols bf16 tile; source rows are
// src_stride elements apart; LDS destination is packed (row stride == cols).
// Descriptor built per CDNA5 ISA section 8 (D# groups 0/1).
__device__ __forceinline__ void tdm_load_2d_bf16(const __bf16* gsrc, __bf16* lds_dst,
                                                 unsigned rows, unsigned cols,
                                                 unsigned src_stride) {
#if __has_builtin(__builtin_amdgcn_tensor_load_to_lds)
    unsigned long long ga = (unsigned long long)(uintptr_t)gsrc;
    unsigned lw = (unsigned)(uintptr_t)lds_dst;     // low 32 bits = LDS byte offset
    v4u g0;
    g0[0] = 1u;                                      // count=1, user descriptor
    g0[1] = lw;                                      // lds_addr
    g0[2] = (unsigned)(ga & 0xFFFFFFFFu);            // global_addr[31:0]
    g0[3] = (unsigned)((ga >> 32) & 0x1FFFFFFu) | (2u << 30);  // addr[56:32] | type=2
    v8i g1;
    g1[0] = (int)(1u << 16);                         // data_size=1 (2B), wg_mask=0
    g1[1] = (int)((src_stride & 0xFFFFu) << 16);     // tensor_dim0[15:0]
    g1[2] = (int)((src_stride >> 16) | ((rows & 0xFFFFu) << 16)); // dim0 hi | dim1 lo
    g1[3] = (int)((rows >> 16) | ((cols & 0xFFFFu) << 16));       // dim1 hi | tile_dim0
    g1[4] = (int)(rows & 0xFFFFu);                   // tile_dim1 | tile_dim2=0
    g1[5] = (int)src_stride;                         // tensor_dim0_stride[31:0]
    g1[6] = 0;                                       // stride0 hi | stride1 lo
    g1[7] = 0;                                       // stride1 hi
    v4i gz = {0, 0, 0, 0};
#if __clang_major__ >= 23
    v8i gz8 = {0, 0, 0, 0, 0, 0, 0, 0};
    __builtin_amdgcn_tensor_load_to_lds(g0, g1, gz, gz, gz8, 0);
#else
    __builtin_amdgcn_tensor_load_to_lds(g0, g1, gz, gz, 0);
#endif
#else
    for (unsigned r = 0; r < rows; ++r)
        for (unsigned i = (threadIdx.x & 31); i < cols; i += 32)
            lds_dst[r * cols + i] = gsrc[(size_t)r * src_stride + i];
#endif
}

__device__ __forceinline__ void tensor_wait_le1() {
#if __has_builtin(__builtin_amdgcn_s_wait_tensorcnt)
    __builtin_amdgcn_s_wait_tensorcnt(1);
#endif
}
__device__ __forceinline__ void tensor_wait_0() {
#if __has_builtin(__builtin_amdgcn_s_wait_tensorcnt)
    __builtin_amdgcn_s_wait_tensorcnt(0);
#endif
}

// ---------------- fused 4-layer MLP: TDM-streamed B tiles + bf16 WMMA ----------------
__global__ void __launch_bounds__(256) k_mlp(
    const __bf16* __restrict__ X,   const __bf16* __restrict__ W1t,
    const __bf16* __restrict__ W2t, const __bf16* __restrict__ W3t,
    const __bf16* __restrict__ W4t,
    const float* __restrict__ fb1, const float* __restrict__ fb2,
    const float* __restrict__ fb3, const float* __restrict__ fb4,
    float* __restrict__ out) {
    // 64 KB total LDS:
    //   [0, 32K):  stage-1 W1t double buffer (2 x 128x32 bf16 = 16 KB)  -- then
    //              reused as per-wave 16x128 activation stripes (32 KB)
    //   [32K,64K): W2t staged by TDM (32 KB), live for stage 2
    __shared__ __align__(16) char smem[65536];
    __bf16* bufW1 = (__bf16*)smem;                 // stage-1 only (aliases slab)
    __bf16* slab  = (__bf16*)smem;                 // stages 1-out .. 4
    __bf16* lW2   = (__bf16*)(smem + 32768);

    const int lane = threadIdx.x & 31;
    const int wave = threadIdx.x >> 5;
    const int lh   = lane >> 4;                    // K-half select (A/B frag layout)
    const int ln   = lane & 15;                    // row (A) / col (B) within tile
    const int rowbase = blockIdx.x * 128 + wave * 16;
    __bf16* ws = &slab[wave * 16 * 128];
    const v8f vzero = {};

    // prologue: TDM W2t (stage 2) and the first W1t K-chunk
    if (wave == 0) {
        tdm_load_2d_bf16(W2t, lW2, 128, 128, 128);
        tdm_load_2d_bf16(W1t, bufW1, 128, 32, KP);
    }

    // ---- stage 1: h1 = lrelu(X @ W1 + fb1)   K=3104, N=128 ----
    // W1t K-chunks (128 x 32 bf16, 8 KB) stream through a TDM double buffer;
    // each chunk is read once from L2 and shared by all 8 waves via LDS.
    v8f acc[8];
    #pragma unroll
    for (int t = 0; t < 8; ++t) acc[t] = vzero;
    const __bf16* arow = X + (size_t)(rowbase + ln) * KP + lh * 16;
    for (int c = 0; c < NCH; ++c) {
        if (c + 1 < NCH) {
            if (wave == 0) {
                tdm_load_2d_bf16(W1t + (c + 1) * 32, bufW1 + ((c + 1) & 1) * (128 * 32),
                                 128, 32, KP);
                tensor_wait_le1();     // chunk c complete (W2t issued first, done first)
            }
        } else {
            if (wave == 0) tensor_wait_0();   // last chunk + W2t all complete
        }
        __syncthreads();               // chunk c visible to all waves
        const __bf16* bw = bufW1 + (c & 1) * (128 * 32);
        v16bf a = ldv16(arow + c * 32);
        if (c + 1 < NCH) __builtin_prefetch(arow + c * 32 + 32, 0, 0);
        #pragma unroll
        for (int t = 0; t < 8; ++t) {
            v16bf b = ldv16(bw + (t * 16 + ln) * 32 + lh * 16);
            acc[t] = __builtin_amdgcn_wmma_f32_16x16x32_bf16(
                false, a, false, b, (short)0, acc[t], false, false);
        }
        __syncthreads();               // safe for TDM to overwrite the other buffer
    }
    // bufW1 region now becomes the activation stripe slab
    #pragma unroll
    for (int t = 0; t < 8; ++t) {
        float bb = fb1[t * 16 + ln];
        #pragma unroll
        for (int i = 0; i < 8; ++i) {
            float v = acc[t][i] + bb;
            v = v > 0.f ? v : 0.01f * v;
            ws[(i + lh * 8) * 128 + t * 16 + ln] = (__bf16)v;
        }
    }
    __syncthreads();

    // ---- stage 2: h2 = lrelu(h1 @ W2 + fb2)   K=128, N=128 (B from TDM LDS) ----
    #pragma unroll
    for (int t = 0; t < 8; ++t) acc[t] = vzero;
    #pragma unroll
    for (int k0 = 0; k0 < 128; k0 += 32) {
        v16bf a = ldv16(&ws[ln * 128 + k0 + lh * 16]);
        #pragma unroll
        for (int t = 0; t < 8; ++t) {
            v16bf b = ldv16(&lW2[(t * 16 + ln) * 128 + k0 + lh * 16]);
            acc[t] = __builtin_amdgcn_wmma_f32_16x16x32_bf16(
                false, a, false, b, (short)0, acc[t], false, false);
        }
    }
    __syncthreads();
    #pragma unroll
    for (int t = 0; t < 8; ++t) {
        float bb = fb2[t * 16 + ln];
        #pragma unroll
        for (int i = 0; i < 8; ++i) {
            float v = acc[t][i] + bb;
            v = v > 0.f ? v : 0.01f * v;
            ws[(i + lh * 8) * 128 + t * 16 + ln] = (__bf16)v;
        }
    }
    __syncthreads();

    // ---- stage 3: h3 = lrelu(h2 @ W3 + fb3)   K=128, N=64 ----
    v8f acc3[4];
    #pragma unroll
    for (int t = 0; t < 4; ++t) acc3[t] = vzero;
    #pragma unroll
    for (int k0 = 0; k0 < 128; k0 += 32) {
        v16bf a = ldv16(&ws[ln * 128 + k0 + lh * 16]);
        #pragma unroll
        for (int t = 0; t < 4; ++t) {
            v16bf b = ldv16(W3t + (size_t)(t * 16 + ln) * 128 + k0 + lh * 16);
            acc3[t] = __builtin_amdgcn_wmma_f32_16x16x32_bf16(
                false, a, false, b, (short)0, acc3[t], false, false);
        }
    }
    __syncthreads();
    #pragma unroll
    for (int t = 0; t < 4; ++t) {
        float bb = fb3[t * 16 + ln];
        #pragma unroll
        for (int i = 0; i < 8; ++i) {
            float v = acc3[t][i] + bb;
            v = v > 0.f ? v : 0.01f * v;
            ws[(i + lh * 8) * 128 + t * 16 + ln] = (__bf16)v;   // 16x64, row stride 128
        }
    }
    __syncthreads();

    // ---- stage 4: out = sigmoid(h3 @ W4 + fb4)   K=64, N=16 (2 real cols) ----
    v8f acc4 = vzero;
    #pragma unroll
    for (int k0 = 0; k0 < 64; k0 += 32) {
        v16bf a = ldv16(&ws[ln * 128 + k0 + lh * 16]);
        v16bf b = ldv16(W4t + (size_t)ln * 64 + k0 + lh * 16);
        acc4 = __builtin_amdgcn_wmma_f32_16x16x32_bf16(
            false, a, false, b, (short)0, acc4, false, false);
    }
    if (ln < 2) {
        float bb = fb4[ln];
        #pragma unroll
        for (int i = 0; i < 8; ++i) {
            float v = acc4[i] + bb;
            v = 1.0f / (1.0f + __expf(-v));
            out[(size_t)(rowbase + i + lh * 8) * 2 + ln] = v;
        }
    }
}

// ---------------- launcher ----------------
extern "C" void kernel_launch(void* const* d_in, const int* in_sizes, int n_in,
                              void* d_out, int out_size, void* d_ws, size_t ws_size,
                              hipStream_t stream) {
    const float* data = (const float*)d_in[0];
    const float* dmat = (const float*)d_in[1];
    const int*   src  = (const int*)d_in[2];
    const int*   dst  = (const int*)d_in[3];
    const float* W0 = (const float*)d_in[4],  *b0 = (const float*)d_in[5];
    const float* G0 = (const float*)d_in[6],  *gb0 = (const float*)d_in[7];
    const float* W1 = (const float*)d_in[8],  *b1 = (const float*)d_in[9];
    const float* G1 = (const float*)d_in[10], *gb1 = (const float*)d_in[11];
    const float* W2 = (const float*)d_in[12], *b2 = (const float*)d_in[13];
    const float* G2 = (const float*)d_in[14], *gb2 = (const float*)d_in[15];
    const float* F1 = (const float*)d_in[16], *fb1 = (const float*)d_in[17];
    const float* F2 = (const float*)d_in[18], *fb2 = (const float*)d_in[19];
    const float* F3 = (const float*)d_in[20], *fb3 = (const float*)d_in[21];
    const float* F4 = (const float*)d_in[22], *fb4 = (const float*)d_in[23];

    char* base = (char*)d_ws;
    size_t off = 0;
    auto alloc = [&](size_t bytes) -> char* {
        char* p = base + off;
        off = (off + bytes + 255) & ~(size_t)255;
        return p;
    };
    __bf16* X   = (__bf16*)alloc((size_t)BB * KP * 2);
    __bf16* W1t = (__bf16*)alloc((size_t)NH1 * KP * 2);
    __bf16* W2t = (__bf16*)alloc((size_t)NH2 * 128 * 2);
    __bf16* W3t = (__bf16*)alloc((size_t)NH3 * 128 * 2);
    __bf16* W4t = (__bf16*)alloc((size_t)NH4P * 64 * 2);
    float*  hA  = (float*)alloc((size_t)BB * HROW * 4);
    float*  hB  = (float*)alloc((size_t)BB * HROW * 4);
    int* cnt    = (int*)alloc(NND * 4);
    int* rowptr = (int*)alloc((NND + 1) * 4);
    int* fill   = (int*)alloc(NND * 4);
    int* cols   = (int*)alloc(EED * 4);

    // CSR build
    k_zero_i32<<<1, 128, 0, stream>>>(cnt, NND);
    k_csr_count<<<(EED + 255) / 256, 256, 0, stream>>>(dst, cnt);
    k_csr_scan<<<1, 32, 0, stream>>>(cnt, rowptr, fill);
    k_csr_fill<<<(EED + 255) / 256, 256, 0, stream>>>(src, dst, fill, cols);

    // 3 gated-RGCN layers
    const int ngrid = (BB * NND + 255) / 256;
    k_rgcn<<<ngrid, 256, 0, stream>>>(data, hA, rowptr, cols, W0, b0, G0, gb0, 1);
    k_rgcn<<<ngrid, 256, 0, stream>>>(hA,   hB, rowptr, cols, W1, b1, G1, gb1, EMBD);
    k_rgcn<<<ngrid, 256, 0, stream>>>(hB,   hA, rowptr, cols, W2, b2, G2, gb2, EMBD);

    // pack activations + weights to bf16 (weights transposed for B-fragment layout)
    k_pack<<<((size_t)BB * KP + 255) / 256, 256, 0, stream>>>(hA, dmat, X);
    k_wt<<<(NH1 * KP + 255) / 256, 256, 0, stream>>>(F1, W1t, XCOL, KP, NH1, NH1);
    k_wt<<<(NH2 * 128 + 255) / 256, 256, 0, stream>>>(F2, W2t, 128, 128, NH2, NH2);
    k_wt<<<(NH3 * 128 + 255) / 256, 256, 0, stream>>>(F3, W3t, 128, 128, NH3, NH3);
    k_wt<<<(NH4P * 64 + 255) / 256, 256, 0, stream>>>(F4, W4t, 64, 64, 2, NH4P);

    // fused WMMA MLP head (TDM double-buffers W1t chunks; W2t TDM-resident in LDS)
    k_mlp<<<BB / 128, 256, 0, stream>>>(X, W1t, W2t, W3t, W4t,
                                        fb1, fb2, fb3, fb4, (float*)d_out);
}
